// Block_13365938225957
// MI455X (gfx1250) — matrix operand order repaired and checked by reading, not compiled, not run on hardware
//
#include <hip/hip_runtime.h>
#include <hip/hip_bf16.h>
#include <math.h>

// ---------------------------------------------------------------------------
// RWKV5 block for MI455X (gfx1250). All big GEMMs run on v_wmma_f32_16x16x32_bf16.
// B=2, T=1024, C=2048, F=7168, H=32, N=64.  M = B*T = 2048.
// GEMM: 128x128 block tile, 8 waves (2x4), BK=64 through LDS with
// register-stage double buffering (global loads overlap WMMA).
// ---------------------------------------------------------------------------

typedef __attribute__((ext_vector_type(16))) __bf16 v16bf;
typedef __attribute__((ext_vector_type(8)))  __bf16 v8bf;
typedef __attribute__((ext_vector_type(8)))  float  v8f;
typedef __attribute__((ext_vector_type(4)))  float  v4f;

#define RW_B 2
#define RW_T 1024
#define RW_C 2048
#define RW_F 7168
#define RW_H 32
#define RW_N 64
#define RW_EPS 1e-5f

#define BM 128
#define BN 128
#define BK 64
#define LDT 72   // LDS stride in halves (144B: 16B aligned, bank-staggered)

enum { EPI_F32 = 0, EPI_SILU = 1, EPI_RESID = 2, EPI_RELU2_BF16 = 3, EPI_FFN = 4 };

template <int EPI>
__global__ __launch_bounds__(256)
void gemm_nt_wmma(const __bf16* __restrict__ A,   // [M,K] bf16
                  const float*  __restrict__ Wt,  // [N,K] fp32 (weight, row=n)
                  void*         __restrict__ Dst, // [M,N] f32 or bf16 per EPI
                  int M, int N, int K,
                  const float* __restrict__ aux0, // EPI_RESID: x ; EPI_FFN: rr
                  const float* __restrict__ aux1) // EPI_FFN: x1
{
    __shared__ __align__(16) __bf16 sA[BM * LDT];
    __shared__ __align__(16) __bf16 sB[BN * LDT];

    const int tid  = threadIdx.x;
    const int lane = tid & 31;
    const int wid  = tid >> 5;
    const int wm   = wid >> 2;   // 0..1
    const int wn   = wid & 3;    // 0..3
    const int m0   = blockIdx.y * BM;
    const int n0   = blockIdx.x * BN;

    v8f acc[4][2];
#pragma unroll
    for (int mi = 0; mi < 4; ++mi)
#pragma unroll
        for (int ni = 0; ni < 2; ++ni)
            acc[mi][ni] = (v8f)(0.0f);

    const int l16  = lane & 15;
    const int hsel = lane >> 4;  // 0/1

    // staging-register tile buffers: 4 chunks of 8 halves / 8 floats each
    v8bf pa[4];
    v4f  pw0[4], pw1[4];
    // chunk geometry: 128 rows x 64 halves = 1024 chunks of 8; 4 per thread
    int crow[4], ccol[4];
#pragma unroll
    for (int it = 0; it < 4; ++it) {
        int chunk = tid + it * 256;
        crow[it] = chunk >> 3;          // 0..127
        ccol[it] = (chunk & 7) << 3;    // 0..56 step 8
    }

    // ---- prologue: load tile kk=0 into registers ----
#pragma unroll
    for (int it = 0; it < 4; ++it) {
        pa[it] = *(const v8bf*)(A + (size_t)(m0 + crow[it]) * K + ccol[it]);
        const float* bp = Wt + (size_t)(n0 + crow[it]) * K + ccol[it];
        pw0[it] = *(const v4f*)bp;
        pw1[it] = *(const v4f*)(bp + 4);
    }

    for (int kk = 0; kk < K; kk += BK) {
        __syncthreads();   // all waves finished reading LDS from previous stage
        // ---- commit staged registers -> LDS (fp32 weights -> bf16) ----
#pragma unroll
        for (int it = 0; it < 4; ++it) {
            *(v8bf*)(sA + crow[it] * LDT + ccol[it]) = pa[it];
            v8bf hv;
            hv[0] = (__bf16)pw0[it][0]; hv[1] = (__bf16)pw0[it][1];
            hv[2] = (__bf16)pw0[it][2]; hv[3] = (__bf16)pw0[it][3];
            hv[4] = (__bf16)pw1[it][0]; hv[5] = (__bf16)pw1[it][1];
            hv[6] = (__bf16)pw1[it][2]; hv[7] = (__bf16)pw1[it][3];
            *(v8bf*)(sB + crow[it] * LDT + ccol[it]) = hv;
        }
        __syncthreads();

        // ---- issue global loads for next stage; they overlap the WMMAs ----
        if (kk + BK < K) {
            int kn = kk + BK;
#pragma unroll
            for (int it = 0; it < 4; ++it) {
                pa[it] = *(const v8bf*)(A + (size_t)(m0 + crow[it]) * K + kn + ccol[it]);
                const float* bp = Wt + (size_t)(n0 + crow[it]) * K + kn + ccol[it];
                pw0[it] = *(const v4f*)bp;
                pw1[it] = *(const v4f*)(bp + 4);
            }
        }
        if (kk + 2 * BK < K) {  // keep L2->L0 path warm two stages ahead
            int kp = kk + 2 * BK;
            __builtin_prefetch(A + (size_t)(m0 + crow[0]) * K + kp + ccol[0], 0, 1);
            __builtin_prefetch(Wt + (size_t)(n0 + crow[0]) * K + kp + ccol[0], 0, 1);
        }

        // ---- two 32-deep WMMA steps on the staged tile ----
#pragma unroll
        for (int ks = 0; ks < BK; ks += 32) {
            v16bf afr[4], bfr[2];
#pragma unroll
            for (int mi = 0; mi < 4; ++mi) {
                int row = wm * 64 + mi * 16 + l16;
                const __bf16* p = sA + row * LDT + ks + hsel * 8;
                union { v16bf v; v8bf h[2]; } u;
                u.h[0] = *(const v8bf*)p;          // K 0..7   (or 8..15)
                u.h[1] = *(const v8bf*)(p + 16);   // K 16..23 (or 24..31)
                afr[mi] = u.v;
            }
#pragma unroll
            for (int ni = 0; ni < 2; ++ni) {
                int row = wn * 32 + ni * 16 + l16;
                const __bf16* p = sB + row * LDT + ks + hsel * 8;
                union { v16bf v; v8bf h[2]; } u;
                u.h[0] = *(const v8bf*)p;          // K 0..7   (or 8..15)
                u.h[1] = *(const v8bf*)(p + 16);   // K 16..23 (or 24..31)
                bfr[ni] = u.v;
            }
#pragma unroll
            for (int mi = 0; mi < 4; ++mi)
#pragma unroll
                for (int ni = 0; ni < 2; ++ni)
                    acc[mi][ni] = __builtin_amdgcn_wmma_f32_16x16x32_bf16(
                        false, afr[mi], false, bfr[ni],
                        (short)0, acc[mi][ni], false, false);
        }
    }

    // ---- epilogue: lane holds N=l16, rows (lane>>4)*8 + i; hoisted indexing ----
    const int rsel = hsel * 8;
#pragma unroll
    for (int mi = 0; mi < 4; ++mi) {
#pragma unroll
        for (int ni = 0; ni < 2; ++ni) {
            const int n = n0 + wn * 32 + ni * 16 + l16;
            size_t idx = (size_t)(m0 + wm * 64 + mi * 16 + rsel) * N + n;
#pragma unroll
            for (int i = 0; i < 8; ++i, idx += (size_t)N) {
                float v = acc[mi][ni][i];
                if (EPI == EPI_F32) {
                    ((float*)Dst)[idx] = v;
                } else if (EPI == EPI_SILU) {
                    ((float*)Dst)[idx] = v / (1.0f + __expf(-v));
                } else if (EPI == EPI_RESID) {
                    ((float*)Dst)[idx] = v + aux0[idx];
                } else if (EPI == EPI_RELU2_BF16) {
                    float t = v > 0.0f ? v : 0.0f;
                    ((__bf16*)Dst)[idx] = (__bf16)(t * t);
                } else { // EPI_FFN: x2 = x1 + sigmoid(rr) * acc
                    float s = 1.0f / (1.0f + __expf(-aux0[idx]));
                    ((float*)Dst)[idx] = aux1[idx] + s * v;
                }
            }
        }
    }
}

// ---------------- LayerNorm over C per token; optional last-token export ----
__global__ __launch_bounds__(256)
void ln_kernel(const float* __restrict__ x, const float* __restrict__ w,
               const float* __restrict__ b, float* __restrict__ y,
               float* __restrict__ last_out, int C, int T)
{
    const int m   = blockIdx.x;
    const int tid = threadIdx.x;
    const float* xr = x + (size_t)m * C;
    float sum = 0.f, sq = 0.f;
    for (int c = tid; c < C; c += 256) { float v = xr[c]; sum += v; sq += v * v; }
    __shared__ float r0[256], r1[256];
    r0[tid] = sum; r1[tid] = sq;
    __syncthreads();
    for (int off = 128; off > 0; off >>= 1) {
        if (tid < off) { r0[tid] += r0[tid + off]; r1[tid] += r1[tid + off]; }
        __syncthreads();
    }
    float mean = r0[0] / (float)C;
    float var  = r1[0] / (float)C - mean * mean;
    float rstd = rsqrtf(var + RW_EPS);
    bool isLast = ((m % T) == (T - 1));
    int  bb = m / T;
    for (int c = tid; c < C; c += 256) {
        float v = (xr[c] - mean) * rstd * w[c] + b[c];
        y[(size_t)m * C + c] = v;
        if (isLast) last_out[(size_t)bb * C + c] = v;
    }
}

// ---------------- token-shift mixes (4-way, time-mix) ----------------------
__global__ __launch_bounds__(256)
void mix4_kernel(const float* __restrict__ xn, const float* __restrict__ tm_shift,
                 const float* __restrict__ tmk, const float* __restrict__ tmv,
                 const float* __restrict__ tmr, const float* __restrict__ tmg,
                 __bf16* __restrict__ xk, __bf16* __restrict__ xv,
                 __bf16* __restrict__ xr, __bf16* __restrict__ xg, int C, int T)
{
    size_t idx = (size_t)blockIdx.x * 256 + threadIdx.x;
    int c = (int)(idx % C);
    size_t m = idx / C;
    int t = (int)(m % T);
    int bb = (int)(m / T);
    float cur = xn[idx];
    float prev = t ? xn[idx - C] : tm_shift[(size_t)bb * C + c];
    float mk = tmk[c], mv = tmv[c], mr = tmr[c], mg = tmg[c];
    xk[idx] = (__bf16)(cur * mk + prev * (1.0f - mk));
    xv[idx] = (__bf16)(cur * mv + prev * (1.0f - mv));
    xr[idx] = (__bf16)(cur * mr + prev * (1.0f - mr));
    xg[idx] = (__bf16)(cur * mg + prev * (1.0f - mg));
}

// ---------------- token-shift mixes (2-way, channel-mix) -------------------
__global__ __launch_bounds__(256)
void mix2_kernel(const float* __restrict__ xn2, const float* __restrict__ cm_shift,
                 const float* __restrict__ cmk, const float* __restrict__ cmr,
                 __bf16* __restrict__ xk2, __bf16* __restrict__ xr2, int C, int T)
{
    size_t idx = (size_t)blockIdx.x * 256 + threadIdx.x;
    int c = (int)(idx % C);
    size_t m = idx / C;
    int t = (int)(m % T);
    int bb = (int)(m / T);
    float cur = xn2[idx];
    float prev = t ? xn2[idx - C] : cm_shift[(size_t)bb * C + c];
    float mk = cmk[c], mr = cmr[c];
    xk2[idx] = (__bf16)(cur * mk + prev * (1.0f - mk));
    xr2[idx] = (__bf16)(cur * mr + prev * (1.0f - mr));
}

// ---------------- WKV recurrence: 1 block per (b,h), thread j owns column j
__global__ __launch_bounds__(64)
void wkv_kernel(const float* __restrict__ r, const float* __restrict__ k,
                const float* __restrict__ v, const float* __restrict__ s0,
                const float* __restrict__ time_decay,
                const float* __restrict__ time_faaaa,
                float* __restrict__ wkv, float* __restrict__ s_out, int T, int C)
{
    const int bh = blockIdx.x;         // b*H + h
    const int b  = bh >> 5;            // H = 32
    const int h  = bh & 31;
    const int j  = threadIdx.x;        // 0..63
    __shared__ float rs[RW_N], ks[RW_N], fa[RW_N], wd[RW_N];
    fa[j] = time_faaaa[h * RW_N + j];
    wd[j] = __expf(-__expf(time_decay[h * RW_N + j]));
    float S[RW_N];
    const float* sp = s0 + (size_t)bh * RW_N * RW_N;
#pragma unroll
    for (int i = 0; i < RW_N; ++i) S[i] = sp[i * RW_N + j];
    size_t base = (size_t)b * T * C + h * RW_N;
    __syncthreads();
    for (int t = 0; t < T; ++t) {
        size_t o = base + (size_t)t * C;
        float rv = r[o + j], kv = k[o + j], vj = v[o + j];
        __syncthreads();               // previous iteration done with rs/ks
        rs[j] = rv; ks[j] = kv;
        __syncthreads();
        float out = 0.f;
#pragma unroll
        for (int i = 0; i < RW_N; ++i) {
            float a = ks[i] * vj;
            out += rs[i] * (fa[i] * a + S[i]);
            S[i] = a + wd[i] * S[i];
        }
        wkv[o + j] = out;
    }
    float* so = s_out + (size_t)bh * RW_N * RW_N;
#pragma unroll
    for (int i = 0; i < RW_N; ++i) so[i * RW_N + j] = S[i];
}

// ---------------- GroupNorm(H groups)/8 * g -> bf16 att --------------------
__global__ __launch_bounds__(256)
void gn_kernel(const float* __restrict__ wkv, const float* __restrict__ g,
               const float* __restrict__ lnx_w, const float* __restrict__ lnx_b,
               __bf16* __restrict__ att, int C)
{
    const int m   = blockIdx.x;
    const int tid = threadIdx.x;
    const int c0  = tid * 8;           // 8 threads per 64-chan group, wave-aligned
    const float* xr = wkv + (size_t)m * C;
    float vals[8], sum = 0.f, sq = 0.f;
#pragma unroll
    for (int u = 0; u < 8; ++u) {
        float v = xr[c0 + u] * 0.125f; // / DIV
        vals[u] = v; sum += v; sq += v * v;
    }
#pragma unroll
    for (int msk = 1; msk < 8; msk <<= 1) {
        sum += __shfl_xor(sum, msk, 32);
        sq  += __shfl_xor(sq,  msk, 32);
    }
    float mean = sum * (1.0f / 64.0f);
    float var  = sq * (1.0f / 64.0f) - mean * mean;
    float rstd = rsqrtf(var + RW_EPS);
#pragma unroll
    for (int u = 0; u < 8; ++u) {
        int c = c0 + u;
        float o = ((vals[u] - mean) * rstd * lnx_w[c] + lnx_b[c]) * g[(size_t)m * C + c];
        att[(size_t)m * C + c] = (__bf16)o;
    }
}

// ---------------------------------------------------------------------------
extern "C" void kernel_launch(void* const* d_in, const int* in_sizes, int n_in,
                              void* d_out, int out_size, void* d_ws, size_t ws_size,
                              hipStream_t stream)
{
    (void)in_sizes; (void)n_in; (void)out_size; (void)ws_size;
    const int B = RW_B, T = RW_T, C = RW_C, F = RW_F, H = RW_H, N = RW_N;
    const int M = B * T;                              // 2048 tokens

    const float* x          = (const float*)d_in[0];
    const float* tm_shift   = (const float*)d_in[1];
    const float* wkv_state  = (const float*)d_in[2];
    const float* cm_shift   = (const float*)d_in[3];
    const float* ln1_w      = (const float*)d_in[4];
    const float* ln1_b      = (const float*)d_in[5];
    const float* ln2_w      = (const float*)d_in[6];
    const float* ln2_b      = (const float*)d_in[7];
    const float* tmk        = (const float*)d_in[8];
    const float* tmv        = (const float*)d_in[9];
    const float* tmr        = (const float*)d_in[10];
    const float* tmg        = (const float*)d_in[11];
    const float* time_decay = (const float*)d_in[12];
    const float* time_faaaa = (const float*)d_in[13];
    const float* Wr         = (const float*)d_in[14];
    const float* Wk         = (const float*)d_in[15];
    const float* Wv         = (const float*)d_in[16];
    const float* Wg         = (const float*)d_in[17];
    const float* Wo         = (const float*)d_in[18];
    const float* lnx_w      = (const float*)d_in[19];
    const float* lnx_b      = (const float*)d_in[20];
    const float* cmk_mix    = (const float*)d_in[21];
    const float* cmr_mix    = (const float*)d_in[22];
    const float* Wck        = (const float*)d_in[23];
    const float* Wcr        = (const float*)d_in[24];
    const float* Wcv        = (const float*)d_in[25];

    // ---- output tuple layout: x2 | xn[:, -1] | s_fin | xn2[:, -1] ----
    float* out_x2      = (float*)d_out;
    float* out_xnlast  = out_x2 + (size_t)M * C;
    float* out_sfin    = out_xnlast + (size_t)B * C;
    float* out_xn2last = out_sfin + (size_t)B * H * N * N;

    // ---- workspace layout (~143 MB, sequential-stream reuse is safe) ----
    char* ws = (char*)d_ws;
    float*  xn   = (float*)(ws + 0);               // 16 MB  (dead after mix4)
    __bf16* xk2b = (__bf16*)(ws + 0);              //   reuse: 8 MB
    __bf16* xr2b = (__bf16*)(ws + 8388608u);       //   reuse: 8 MB
    __bf16* xrb  = (__bf16*)(ws + 16777216u);      // 8 MB each
    __bf16* xkb  = (__bf16*)(ws + 25165824u);
    __bf16* xvb  = (__bf16*)(ws + 33554432u);
    __bf16* xgb  = (__bf16*)(ws + 41943040u);      // region dead after g GEMM
    float*  x1   = (float*)(ws + 16777216u);       //   reuse: 16 MB
    float*  xn2  = (float*)(ws + 33554432u);       //   reuse: 16 MB
    float*  rf   = (float*)(ws + 50331648u);       // 16 MB each
    float*  kf   = (float*)(ws + 67108864u);
    float*  vf   = (float*)(ws + 83886080u);       // region dead after wkv
    __bf16* kkb  = (__bf16*)(ws + 50331648u);      //   reuse: 28 MB (M*F bf16)
    float*  rr   = (float*)(ws + 79691776u);       //   reuse: 16 MB
    float*  gf   = (float*)(ws + 100663296u);      // 16 MB
    float*  wkv  = (float*)(ws + 117440512u);      // 16 MB
    __bf16* attb = (__bf16*)(ws + 134217728u);     // 8 MB

    const dim3 blk(256);
    const dim3 gCC(C / BN, M / BM);   // 16 x 16
    const dim3 gCF(F / BN, M / BM);   // 56 x 16
    const int  nElem = (M * C) / 256; // 16384 blocks for elementwise passes

    // ---------- time-mix ----------
    ln_kernel<<<M, blk, 0, stream>>>(x, ln1_w, ln1_b, xn, out_xnlast, C, T);
    mix4_kernel<<<nElem, blk, 0, stream>>>(xn, tm_shift, tmk, tmv, tmr, tmg,
                                           xkb, xvb, xrb, xgb, C, T);
    gemm_nt_wmma<EPI_F32><<<gCC, blk, 0, stream>>>(xrb, Wr, rf, M, C, C, nullptr, nullptr);
    gemm_nt_wmma<EPI_F32><<<gCC, blk, 0, stream>>>(xkb, Wk, kf, M, C, C, nullptr, nullptr);
    gemm_nt_wmma<EPI_F32><<<gCC, blk, 0, stream>>>(xvb, Wv, vf, M, C, C, nullptr, nullptr);
    gemm_nt_wmma<EPI_SILU><<<gCC, blk, 0, stream>>>(xgb, Wg, gf, M, C, C, nullptr, nullptr);

    wkv_kernel<<<B * H, dim3(64), 0, stream>>>(rf, kf, vf, wkv_state,
                                               time_decay, time_faaaa,
                                               wkv, out_sfin, T, C);
    gn_kernel<<<M, blk, 0, stream>>>(wkv, gf, lnx_w, lnx_b, attb, C);
    gemm_nt_wmma<EPI_RESID><<<gCC, blk, 0, stream>>>(attb, Wo, x1, M, C, C, x, nullptr);

    // ---------- channel-mix ----------
    ln_kernel<<<M, blk, 0, stream>>>(x1, ln2_w, ln2_b, xn2, out_xn2last, C, T);
    mix2_kernel<<<nElem, blk, 0, stream>>>(xn2, cm_shift, cmk_mix, cmr_mix,
                                           xk2b, xr2b, C, T);
    gemm_nt_wmma<EPI_RELU2_BF16><<<gCF, blk, 0, stream>>>(xk2b, Wck, kkb, M, F, C,
                                                          nullptr, nullptr);
    gemm_nt_wmma<EPI_F32><<<gCC, blk, 0, stream>>>(xr2b, Wcr, rr, M, C, C,
                                                   nullptr, nullptr);
    gemm_nt_wmma<EPI_FFN><<<gCC, blk, 0, stream>>>(kkb, Wcv, out_x2, M, C, F,
                                                   rr, x1);
}
